// PCANet_60670708023690
// MI455X (gfx1250) — matrix operand order repaired and checked by reading, not compiled, not run on hardware
//
#include <hip/hip_runtime.h>
#include <hip/hip_bf16.h>

// CDNA5 wave32 WMMA types
typedef __attribute__((ext_vector_type(16))) __bf16 v16bf;
typedef __attribute__((ext_vector_type(8)))  float  v8f;
// TDM descriptor groups
typedef __attribute__((ext_vector_type(4))) unsigned int v4u;
typedef __attribute__((ext_vector_type(8))) int          v8i;
typedef __attribute__((ext_vector_type(4))) int          v4i;

// conv tile geometry: (16+6) rows x 136 cols (128 + 6 halo, +2 pad)
#define T1_W   136
#define T1_H   22
#define T1_CH  (T1_H * T1_W)           // 2992 elements per channel plane
#define T1_SZ  (3 * T1_CH)             // 8976 bf16 (conv1: 3 input channels)

// LDS element offset of tap k inside conv1 tile (k = c*49 + ky*7 + kx), -1 pad
__device__ __forceinline__ int off_conv1(int k) {
  return (k < 147) ? ((k / 49) * T1_CH + ((k % 49) / 7) * T1_W + ((k % 49) % 7))
                   : -1;
}
// same for conv2 (single 22x136 plane)
__device__ __forceinline__ int off_conv2(int k) {
  return (k < 49) ? ((k / 7) * T1_W + (k % 7)) : -1;
}

// ---------------------------------------------------------------------------
// Kernel 1: conv1 as implicit GEMM on v_wmma_f32_16x16x32_bf16.
// One WG per (image, 16-row strip). Input tile staged in LDS (f32->bf16 on the
// way in), B-fragment gather is branch-free ds_load_u16 + cndmask.
// ---------------------------------------------------------------------------
__global__ __launch_bounds__(256) void conv1_wmma(const float* __restrict__ x,
                                                  const float* __restrict__ w1,
                                                  __bf16* __restrict__ out1) {
  __shared__ __bf16 xt[T1_SZ];

  const int tid  = threadIdx.x;
  const int lane = tid & 31;
  const int wid  = tid >> 5;
  const int hi   = (lane >= 16) ? 1 : 0;
  const int m    = lane & 15;                    // A-row = out channel

  const int img = blockIdx.x >> 3;               // 32 images
  const int s   = blockIdx.x & 7;                // 8 strips of 16 rows

  // ---- cooperative halo'd tile load: 3 x 22 x 136, zero 'same' padding ----
  const float* xsrc = x + (size_t)img * 3 * 16384;
  for (int idx = tid; idx < T1_SZ; idx += 256) {
    int c   = idx / T1_CH;
    int rem = idx - c * T1_CH;
    int ly  = rem / T1_W, lx = rem - ly * T1_W;
    int gy  = 16 * s - 3 + ly;
    int gx  = lx - 3;
    float v = 0.f;
    if (gy >= 0 && gy < 128 && gx >= 0 && gx < 128)
      v = xsrc[(c << 14) + gy * 128 + gx];
    xt[idx] = (__bf16)v;
  }

  // ---- A fragments (weights): branch-free clamped loads, w1_flat = m*147+k
  v16bf a[5];
#pragma unroll
  for (int ch = 0; ch < 5; ++ch) {
#pragma unroll
    for (int e = 0; e < 16; ++e) {
      int k0 = ch * 32 + ((e < 8) ? e : e + 8);  // A lane/K map (lo half)
      int k  = k0 + (hi ? 8 : 0);
      bool valid = (m < 8) && (k < 147);
      float v = w1[valid ? (m * 147 + k) : 0];
      a[ch][e] = (__bf16)(valid ? v : 0.f);
    }
  }
  __syncthreads();

  const int xcol = lane & 15;                    // B column = pixel within seg
#pragma unroll 1
  for (int i = 0; i < 16; ++i) {
    int t    = wid + 8 * i;                      // 128 tiles per strip
    int row  = t >> 3;                           // 0..15 within strip
    int seg  = (t & 7) << 4;
    int base = row * T1_W + seg + xcol;          // LDS base of this pixel

    v8f acc = {};
#pragma unroll
    for (int chk = 0; chk < 5; ++chk) {
      v16bf b;
#pragma unroll
      for (int e = 0; e < 16; ++e) {
        int o0 = off_conv1(chk * 32 + ((e < 8) ? e : e + 8));        // folds to literal
        int o1 = off_conv1(chk * 32 + ((e < 8) ? e : e + 8) + 8);   // folds to literal
        int o  = hi ? o1 : o0;                   // one v_cndmask
        __bf16 v = xt[base + (o >= 0 ? o : 0)];  // unconditional ds_load_u16
        b[e] = (o >= 0) ? v : (__bf16)0.f;
      }
      acc = __builtin_amdgcn_wmma_f32_16x16x32_bf16(false, a[chk], false, b,
                                                    (short)0, acc, false, false);
    }
    // D layout: lanes 0..15 hold pixel = lane, channels 0..7 in acc[0..7]
    if (lane < 16) {
      int y  = 16 * s + row;
      int xx = seg + lane;
#pragma unroll
      for (int r = 0; r < 8; ++r)
        out1[((img * 8 + r) * 128 + y) * 128 + xx] = (__bf16)acc[r];
    }
  }
}

// ---------------------------------------------------------------------------
// Kernel 2: fused conv2 + heaviside bit-pack + per-block histogram.
// One WG per (image, l1-channel, 16-row strip). The out1 tile is DMA'd into
// LDS by the Tensor Data Mover (tensor_load_to_lds): 2D bf16 tile, LDS row
// pitch 136 expressed via D# pad (64 DW interval / 4 DW amount), halo kept
// zero by pre-clearing LDS and clipping the transfer to valid rows.
// ---------------------------------------------------------------------------
__global__ __launch_bounds__(256) void conv2_hist_wmma(const __bf16* __restrict__ out1,
                                                       const float* __restrict__ w2,
                                                       float* __restrict__ out) {
  __shared__ __bf16 tile[T1_CH];      // 22 x 136
  __shared__ float  hist[32 * 64];    // 32 blocks x 64 bins = 8KB

  const int tid  = threadIdx.x;
  const int lane = tid & 31;
  const int wid  = tid >> 5;
  const int hi   = (lane >= 16) ? 1 : 0;

  const int b = blockIdx.x;           // 2048 WGs: 32 img * 8 ch * 8 strips
  const int n = b >> 6;
  const int c = (b >> 3) & 7;
  const int s = b & 7;                // strip: rows [16s, 16s+16)

  // zero histogram + zero tile (halo must stay zero for 'same' padding)
#pragma unroll
  for (int j = 0; j < 8; ++j) hist[tid + 256 * j] = 0.f;
  unsigned int* tz = (unsigned int*)tile;
  for (int idx = tid; idx < (T1_CH / 2); idx += 256) tz[idx] = 0u;

  // A fragments from w2: M = 16 (8 l2 channels valid), K = 64 (49 valid)
  v16bf a[2];
  const int m = lane & 15;
#pragma unroll
  for (int ch = 0; ch < 2; ++ch) {
#pragma unroll
    for (int e = 0; e < 16; ++e) {
      int k0 = ch * 32 + ((e < 8) ? e : e + 8);
      int k  = k0 + (hi ? 8 : 0);
      bool valid = (m < 8) && (k < 49);
      float v = w2[valid ? (m * 49 + k) : 0];
      a[ch][e] = (__bf16)(valid ? v : 0.f);
    }
  }
  __syncthreads();     // zero-fill visible before TDM writes

  // ---- Tensor Data Mover: DMA the clipped 2D tile (rows x 128 bf16) ----
  const __bf16* src = out1 + (size_t)(n * 8 + c) * 16384;
  {
    int y0  = 16 * s - 3;
    int cy0 = (y0 < 0) ? 0 : y0;
    int cy1 = 16 * s + 19; if (cy1 > 128) cy1 = 128;
    int rows = cy1 - cy0;
    if (wid == 0) {    // uniform branch: exactly one wave issues the TDM
      unsigned long long ga =
          (unsigned long long)(size_t)(const void*)(src + (size_t)cy0 * 128);
      unsigned int lds =
          (unsigned int)(size_t)(const void*)tile +
          (unsigned int)(((cy0 - y0) * T1_W + 3) * 2);
      v4u g0;
      g0[0] = 1u;                                    // count=1, user descriptor
      g0[1] = lds;                                   // lds_addr (bytes)
      g0[2] = (unsigned int)(ga & 0xFFFFFFFFull);    // global_addr[31:0]
      g0[3] = (unsigned int)(ga >> 32) | (2u << 30); // global_addr[56:32]|type=2
      v8i g1;
      g1[0] = (1 << 16) | (1 << 20) | (5 << 22) | (3 << 25);
              // data_size=1 (2B), pad_enable, pad_interval=5 (64 DW = 1 row),
              // pad_amount=3 (4 DW = 8 bf16 -> LDS pitch 136)
      g1[1] = 128 << 16;                             // tensor_dim0 = 128
      g1[2] = 128 << 16;                             // tensor_dim1 = 128
      g1[3] = 128 << 16;                             // tile_dim0   = 128
      g1[4] = rows & 0xFFFF;                         // tile_dim1   = rows (2D)
      g1[5] = 128;                                   // tensor_dim0_stride = 128
      g1[6] = 0;
      g1[7] = 0;
      v4i gz = (v4i){0, 0, 0, 0};
#if defined(__clang_major__) && (__clang_major__ >= 23)
      v8i gz8 = (v8i){0, 0, 0, 0, 0, 0, 0, 0};
      __builtin_amdgcn_tensor_load_to_lds(g0, g1, gz, gz, gz8, 0);
#else
      __builtin_amdgcn_tensor_load_to_lds(g0, g1, gz, gz, 0);
#endif
      __builtin_amdgcn_s_wait_tensorcnt(0);
    }
  }
  __syncthreads();     // tile ready for all waves

  const int out_base = (n * 8 + c) * 16384 + s * 2048;
  const int xcol = lane & 15;

#pragma unroll 1
  for (int i = 0; i < 16; ++i) {
    int t    = wid + 8 * i;                   // 128 tiles per strip
    int row  = t >> 3;                        // 0..15 within strip
    int seg  = (t & 7) << 4;
    int x    = seg + xcol;                    // B column = pixel x
    int base = row * T1_W + x;

    v8f acc = {};
#pragma unroll
    for (int chk = 0; chk < 2; ++chk) {
      v16bf bf;
#pragma unroll
      for (int e = 0; e < 16; ++e) {
        int o0 = off_conv2(chk * 32 + ((e < 8) ? e : e + 8));     // folds to literal
        int o1 = off_conv2(chk * 32 + ((e < 8) ? e : e + 8) + 8);
        int o  = hi ? o1 : o0;
        __bf16 v = tile[base + (o >= 0 ? o : 0)];
        bf[e] = (o >= 0) ? v : (__bf16)0.f;
      }
      acc = __builtin_amdgcn_wmma_f32_16x16x32_bf16(false, a[chk], false, bf,
                                                    (short)0, acc, false, false);
    }

    // lanes 0..15: all 8 l2-channel results for pixel (row, x) are lane-local
    if (lane < 16) {
      int code = 0;
#pragma unroll
      for (int r = 0; r < 8; ++r) code |= (acc[r] > 0.f) ? (1 << r) : 0;
      int bucket = code >> 2;                       // // DIV (=4)
      int hb = ((row >> 3) << 4) + (x >> 3);        // local block id (2x16)
      atomicAdd(&hist[hb * 64 + bucket], 1.0f);     // ds_add_f32
    }
  }
  __syncthreads();

  // coalesced 8KB store; layout matches reference (num, ch, by, bx, bin) flatten
#pragma unroll
  for (int j = 0; j < 8; ++j)
    out[out_base + tid + 256 * j] = hist[tid + 256 * j];
}

extern "C" void kernel_launch(void* const* d_in, const int* in_sizes, int n_in,
                              void* d_out, int out_size, void* d_ws, size_t ws_size,
                              hipStream_t stream) {
  const float* x  = (const float*)d_in[0];   // (32,3,128,128) f32
  const float* w1 = (const float*)d_in[1];   // (8,3,7,7) f32
  const float* w2 = (const float*)d_in[2];   // (8,1,7,7) f32
  __bf16* out1 = (__bf16*)d_ws;              // staged conv1 output: 8.4 MB bf16

  conv1_wmma<<<256, 256, 0, stream>>>(x, w1, out1);
  conv2_hist_wmma<<<2048, 256, 0, stream>>>(out1, w2, (float*)d_out);
}
